// MultiHeadAttention_70832600646398
// MI455X (gfx1250) — matrix operand order repaired and checked by reading, not compiled, not run on hardware
//
#include <hip/hip_runtime.h>
#include <hip/hip_bf16.h>

#define NHEAD  16
#define DMODEL 1024
#define DK     64
#define BATCH  8
#define SEQ    1024
#define MROWS  (BATCH * SEQ)   // 8192

typedef __attribute__((ext_vector_type(16))) _Float16 v16h;
typedef __attribute__((ext_vector_type(8)))  _Float16 v8h;
typedef __attribute__((ext_vector_type(4)))  _Float16 v4h;
typedef __attribute__((ext_vector_type(8)))  float    v8f;
typedef __attribute__((ext_vector_type(4)))  float    v4f;

union FragU { v16h v; v8h h[2]; };

// ---------------------------------------------------------------------------
// CDNA5 async global->LDS copy (GLOBAL_LOAD_ASYNC_TO_LDS_B128, ASYNCcnt)
// ---------------------------------------------------------------------------
__device__ __forceinline__ void async_copy_b128(void* lds, const void* gaddr) {
  asm volatile("global_load_async_to_lds_b128 %0, %1, off"
               :: "v"((unsigned)(size_t)lds), "v"(gaddr)
               : "memory");
}
__device__ __forceinline__ void wait_async() {
  asm volatile("s_wait_asynccnt 0x0" ::: "memory");
}

// ---------------------------------------------------------------------------
// WMMA helpers (V_WMMA_F32_16X16X32_F16, wave32)
// ---------------------------------------------------------------------------
__device__ __forceinline__ v8f wmma_f16(v16h a, v16h b, v8f c) {
  return __builtin_amdgcn_wmma_f32_16x16x32_f16(false, a, false, b,
                                                (short)0, c, false, false);
}

// A fragment 16x32 f16: lane L<16 -> M=L, K={0..7,16..23}; L>=16 -> K={8..15,24..31}
__device__ __forceinline__ v16h frag_a(const _Float16* sm, int lda, int m0, int k0) {
  const int lane = threadIdx.x & 31;
  const int r  = m0 + (lane & 15);
  const int kb = k0 + ((lane >> 4) << 3);
  FragU f;
  f.h[0] = *(const v8h*)(sm + r * lda + kb);
  f.h[1] = *(const v8h*)(sm + r * lda + kb + 16);
  return f.v;
}

// B fragment 32x16 f16 from LDS [N][K]: lane L<16 -> N=L,K={0..7,16..23}; L>=16 -> K={8..15,24..31}
__device__ __forceinline__ v16h frag_b(const _Float16* sm, int ldb, int n0, int k0) {
  const int lane = threadIdx.x & 31;
  const int c  = n0 + (lane & 15);
  const int kb = k0 + ((lane >> 4) << 3);
  FragU f;
  f.h[0] = *(const v8h*)(sm + c * ldb + kb);
  f.h[1] = *(const v8h*)(sm + c * ldb + kb + 16);
  return f.v;
}

// ---------------------------------------------------------------------------
// Kernel 1: fused QKV projection (software-pipelined K loop).
//   acc[m,n] = sum_k A[m,k]*W[n,k] + bias[n]
//   zz=0 -> qh head-major  [B][H][S][DK] f16
//   zz=1 -> kh head-major  [B][H][S][DK] f16
//   zz=2 -> vt transposed  [B][H][DK][S] f16
// grid (D/64, M/128, 3), block 256 (8 waves, 4x2, 32x32 each)
// ---------------------------------------------------------------------------
__global__ __launch_bounds__(256) void proj_qkv_kernel(
    const float* __restrict__ q, const float* __restrict__ k, const float* __restrict__ v,
    const float* __restrict__ wq, const float* __restrict__ wk, const float* __restrict__ wv,
    const float* __restrict__ bq, const float* __restrict__ bk, const float* __restrict__ bv,
    _Float16* __restrict__ oq, _Float16* __restrict__ ok, _Float16* __restrict__ ov)
{
  constexpr int BM = 128, BN = 64, BK = 32, LDA = 40, LDB = 40;
  __shared__ __align__(16) _Float16 sA[BM * LDA];
  __shared__ __align__(16) _Float16 sB[BN * LDB];

  const int zz = blockIdx.z;
  const float* A    = (zz == 0) ? q  : (zz == 1) ? k  : v;
  const float* W    = (zz == 0) ? wq : (zz == 1) ? wk : wv;
  const float* bias = (zz == 0) ? bq : (zz == 1) ? bk : bv;
  _Float16*    O    = (zz == 0) ? oq : (zz == 1) ? ok : ov;

  const int m0 = blockIdx.y * BM;
  const int n0 = blockIdx.x * BN;
  const int tid  = threadIdx.x;
  const int wave = tid >> 5;
  const int wm = (wave >> 1) * 32;
  const int wn = (wave & 1) * 32;

  v4f ra[4], rb[2];
  auto load_tiles = [&](int k0) {
#pragma unroll
    for (int i = 0; i < 4; ++i) {          // A tile 128x32: 1024 float4 chunks
      int ch = tid + i * 256;
      int r = ch >> 3, c = (ch & 7) * 4;
      ra[i] = *(const v4f*)(A + (size_t)(m0 + r) * DMODEL + k0 + c);
    }
#pragma unroll
    for (int i = 0; i < 2; ++i) {          // B tile 64x32: 512 float4 chunks
      int ch = tid + i * 256;
      int r = ch >> 3, c = (ch & 7) * 4;
      rb[i] = *(const v4f*)(W + (size_t)(n0 + r) * DMODEL + k0 + c);
    }
  };
  auto store_tiles = [&]() {
#pragma unroll
    for (int i = 0; i < 4; ++i) {
      int ch = tid + i * 256;
      int r = ch >> 3, c = (ch & 7) * 4;
      *(v4h*)(sA + r * LDA + c) = __builtin_convertvector(ra[i], v4h);
    }
#pragma unroll
    for (int i = 0; i < 2; ++i) {
      int ch = tid + i * 256;
      int r = ch >> 3, c = (ch & 7) * 4;
      *(v4h*)(sB + r * LDB + c) = __builtin_convertvector(rb[i], v4h);
    }
  };

  v8f acc[2][2] = {};
  load_tiles(0);
  for (int k0 = 0; k0 < DMODEL; k0 += BK) {
    store_tiles();
    __syncthreads();
    if (k0 + BK < DMODEL) load_tiles(k0 + BK);   // in flight during WMMAs
    v16h a0 = frag_a(sA, LDA, wm, 0);
    v16h a1 = frag_a(sA, LDA, wm + 16, 0);
    v16h b0 = frag_b(sB, LDB, wn, 0);
    v16h b1 = frag_b(sB, LDB, wn + 16, 0);
    acc[0][0] = wmma_f16(a0, b0, acc[0][0]);
    acc[0][1] = wmma_f16(a0, b1, acc[0][1]);
    acc[1][0] = wmma_f16(a1, b0, acc[1][0]);
    acc[1][1] = wmma_f16(a1, b1, acc[1][1]);
    __syncthreads();
  }

  const int lane = tid & 31;
#pragma unroll
  for (int im = 0; im < 2; ++im)
#pragma unroll
    for (int in = 0; in < 2; ++in) {
      int n  = n0 + wn + in * 16 + (lane & 15);
      int mb = m0 + wm + im * 16 + ((lane >> 4) << 3);
      float bb = bias[n];
      int h = n >> 6, d = n & 63;
      int b = mb >> 10, s = mb & (SEQ - 1);     // BM=128 never crosses a batch
      if (zz == 2) {
        v8h pk;
#pragma unroll
        for (int r = 0; r < 8; ++r) pk[r] = (_Float16)(acc[im][in][r] + bb);
        *(v8h*)(O + (((size_t)b * NHEAD + h) * DK + d) * SEQ + s) = pk;
      } else {
#pragma unroll
        for (int r = 0; r < 8; ++r)
          O[(((size_t)b * NHEAD + h) * SEQ + s + r) * DK + d] =
              (_Float16)(acc[im][in][r] + bb);
      }
    }
}

// ---------------------------------------------------------------------------
// Kernel 2: scores[b,h,q,k] = (qh . kh)/8, masked.  qh/kh head-major -> the
// A/B tiles are contiguous blocks, copied with global_load_async_to_lds_b128.
// Epilogue staged through LDS for coalesced float4 stores of the 512MB attn.
// grid (S/64, S/128, B*H), block 256
// ---------------------------------------------------------------------------
__global__ __launch_bounds__(256) void scores_kernel(
    const _Float16* __restrict__ qh, const _Float16* __restrict__ kh,
    const unsigned char* __restrict__ mask, float* __restrict__ attn)
{
  constexpr int BM = 128, BN = 64, KK = 64, LDA = 72, LDB = 72, LDC = 36;
  __shared__ __align__(16) _Float16 sA[BM * LDA];   // 18432 B
  __shared__ __align__(16) _Float16 sB[BN * LDB];   //  9216 B
  float* stage = (float*)sA;                        // 128 x 36 f32 == 18432 B

  const int bh = blockIdx.z;
  const int b = bh >> 4;
  const int m0 = blockIdx.y * BM, n0 = blockIdx.x * BN;
  const int tid = threadIdx.x;

  const _Float16* Aq = qh + ((size_t)bh * SEQ + m0) * DK;  // contiguous 128x64
  const _Float16* Bk = kh + ((size_t)bh * SEQ + n0) * DK;  // contiguous  64x64
#pragma unroll
  for (int i = 0; i < 4; ++i) {            // A: 1024 chunks of 8 halves
    int ch = tid + i * 256;
    int r = ch >> 3, c = (ch & 7) * 8;
    async_copy_b128(sA + r * LDA + c, Aq + (size_t)ch * 8);
  }
#pragma unroll
  for (int i = 0; i < 2; ++i) {            // B: 512 chunks of 8 halves
    int ch = tid + i * 256;
    int r = ch >> 3, c = (ch & 7) * 8;
    async_copy_b128(sB + r * LDB + c, Bk + (size_t)ch * 8);
  }
  wait_async();
  __syncthreads();

  const int wave = tid >> 5;
  const int wm = (wave >> 1) * 32, wn = (wave & 1) * 32;
  const int lane = tid & 31;
  v8f acc[2][2] = {};
#pragma unroll
  for (int kk = 0; kk < KK; kk += 32) {
    v16h a0 = frag_a(sA, LDA, wm, kk);
    v16h a1 = frag_a(sA, LDA, wm + 16, kk);
    v16h b0 = frag_b(sB, LDB, wn, kk);
    v16h b1 = frag_b(sB, LDB, wn + 16, kk);
    acc[0][0] = wmma_f16(a0, b0, acc[0][0]);
    acc[0][1] = wmma_f16(a0, b1, acc[0][1]);
    acc[1][0] = wmma_f16(a1, b0, acc[1][0]);
    acc[1][1] = wmma_f16(a1, b1, acc[1][1]);
  }
  __syncthreads();   // all frag reads done before sA is reused as stage

  float* outp = attn + ((size_t)bh * SEQ + m0) * SEQ + n0;
  const unsigned char* mbase = mask + (size_t)b * SEQ * SEQ + n0;
#pragma unroll
  for (int p = 0; p < 2; ++p) {
    if (wn == p * 32) {                    // wave-uniform branch
#pragma unroll
      for (int im = 0; im < 2; ++im)
#pragma unroll
        for (int in = 0; in < 2; ++in) {
          int nl = in * 16 + (lane & 15);
          int mb = wm + im * 16 + ((lane >> 4) << 3);
#pragma unroll
          for (int r = 0; r < 8; ++r)
            stage[(mb + r) * LDC + nl] = acc[im][in][r] * 0.125f;
        }
    }
    __syncthreads();
#pragma unroll
    for (int i = 0; i < 4; ++i) {          // 128 rows x 8 float4 chunks
      int ch = tid + i * 256;
      int r = ch >> 3, c = (ch & 7) * 4;
      v4f val = *(const v4f*)(stage + r * LDC + c);
      const unsigned char* mp = mbase + (size_t)(m0 + r) * SEQ + p * 32 + c;
      if (mp[0]) val.x = -1.0e9f;
      if (mp[1]) val.y = -1.0e9f;
      if (mp[2]) val.z = -1.0e9f;
      if (mp[3]) val.w = -1.0e9f;
      *(v4f*)(outp + (size_t)r * SEQ + p * 32 + c) = val;
    }
    __syncthreads();
  }
}

// ---------------------------------------------------------------------------
// Kernel 3: row softmax in place over last dim (1024). grid B*H*S, block 256
// ---------------------------------------------------------------------------
__global__ __launch_bounds__(256) void softmax_kernel(float* __restrict__ attn)
{
  const size_t row = blockIdx.x;
  float* p = attn + row * SEQ;
  const int tid = threadIdx.x;

  float vals[4];
  float vmax = -3.0e38f;
#pragma unroll
  for (int i = 0; i < 4; ++i) { vals[i] = p[tid + i * 256]; vmax = fmaxf(vmax, vals[i]); }
#pragma unroll
  for (int off = 16; off > 0; off >>= 1) vmax = fmaxf(vmax, __shfl_xor(vmax, off, 32));

  __shared__ float redm[8];
  __shared__ float reds[8];
  if ((tid & 31) == 0) redm[tid >> 5] = vmax;
  __syncthreads();
  float m = fmaxf(fmaxf(fmaxf(redm[0], redm[1]), fmaxf(redm[2], redm[3])),
                  fmaxf(fmaxf(redm[4], redm[5]), fmaxf(redm[6], redm[7])));

  float sum = 0.f;
#pragma unroll
  for (int i = 0; i < 4; ++i) { vals[i] = __expf(vals[i] - m); sum += vals[i]; }
#pragma unroll
  for (int off = 16; off > 0; off >>= 1) sum += __shfl_xor(sum, off, 32);
  if ((tid & 31) == 0) reds[tid >> 5] = sum;
  __syncthreads();
  float tot = (reds[0] + reds[1]) + (reds[2] + reds[3]) +
              (reds[4] + reds[5]) + (reds[6] + reds[7]);
  float inv = 1.0f / tot;
#pragma unroll
  for (int i = 0; i < 4; ++i) p[tid + i * 256] = vals[i] * inv;
}

// ---------------------------------------------------------------------------
// Kernel 4: out[b,q,h*64+d] = sum_k attn[b,h,q,k] * vt[b,h,d,k]
// A tile register-pipelined; B tile ping-pong async-copied (overlaps WMMAs).
// grid (1, S/128, B*H), block 256
// ---------------------------------------------------------------------------
__global__ __launch_bounds__(256) void attnv_kernel(
    const float* __restrict__ attn, const _Float16* __restrict__ vt,
    float* __restrict__ outp)
{
  constexpr int BM = 128, BK = 32, LDA = 40, LDB = 40;
  __shared__ __align__(16) _Float16 sA[BM * LDA];
  __shared__ __align__(16) _Float16 sB[2][64 * LDB];

  const int bh = blockIdx.z, b = bh >> 4, h = bh & 15;
  const int m0 = blockIdx.y * BM;
  const int tid  = threadIdx.x;
  const int wave = tid >> 5;
  const int wm = (wave >> 1) * 32, wn = (wave & 1) * 32;

  const float*    Ap = attn + ((size_t)bh * SEQ + m0) * SEQ;
  const _Float16* Vp = vt + (size_t)bh * DK * SEQ;

  v4f ra[4];
  auto loadA = [&](int k0) {
#pragma unroll
    for (int i = 0; i < 4; ++i) {          // A tile 128x32 f32
      int ch = tid + i * 256;
      int r = ch >> 3, c = (ch & 7) * 4;
      ra[i] = *(const v4f*)(Ap + (size_t)r * SEQ + k0 + c);
    }
  };
  auto issueB = [&](int k0, int buf) {     // B tile [64 d][32 k]: 256 chunks
    int d = tid >> 2, c = (tid & 3) * 8;
    async_copy_b128(&sB[buf][d * LDB + c], Vp + (size_t)d * SEQ + k0 + c);
  };

  v8f acc[2][2] = {};
  constexpr int NIT = SEQ / BK;            // 32
  loadA(0);
  issueB(0, 0);
  for (int it = 0; it < NIT; ++it) {
#pragma unroll
    for (int i = 0; i < 4; ++i) {          // cvt + store A regs -> LDS
      int ch = tid + i * 256;
      int r = ch >> 3, c = (ch & 7) * 4;
      *(v4h*)(sA + r * LDA + c) = __builtin_convertvector(ra[i], v4h);
    }
    wait_async();
    __syncthreads();
    if (it + 1 < NIT) {                    // next tiles in flight during WMMAs
      loadA((it + 1) * BK);
      issueB((it + 1) * BK, (it + 1) & 1);
    }
    const _Float16* sBc = sB[it & 1];
    v16h a0 = frag_a(sA, LDA, wm, 0);
    v16h a1 = frag_a(sA, LDA, wm + 16, 0);
    v16h b0 = frag_b(sBc, LDB, wn, 0);
    v16h b1 = frag_b(sBc, LDB, wn + 16, 0);
    acc[0][0] = wmma_f16(a0, b0, acc[0][0]);
    acc[0][1] = wmma_f16(a0, b1, acc[0][1]);
    acc[1][0] = wmma_f16(a1, b0, acc[1][0]);
    acc[1][1] = wmma_f16(a1, b1, acc[1][1]);
    __syncthreads();
  }

  const int lane = tid & 31;
#pragma unroll
  for (int im = 0; im < 2; ++im)
#pragma unroll
    for (int in = 0; in < 2; ++in) {
      int n  = wn + in * 16 + (lane & 15);
      int mb = m0 + wm + im * 16 + ((lane >> 4) << 3);
#pragma unroll
      for (int r = 0; r < 8; ++r)
        outp[(size_t)(b * SEQ + mb + r) * DMODEL + h * DK + n] = acc[im][in][r];
    }
}

// ---------------------------------------------------------------------------
// Kernel 5: X@Wfc.T + b_fc + residual -> f32 (software-pipelined).
// grid (D/64, M/128), block 256
// ---------------------------------------------------------------------------
__global__ __launch_bounds__(256) void fc_kernel(
    const float* __restrict__ X, const float* __restrict__ W,
    const float* __restrict__ bias, const float* __restrict__ residual,
    float* __restrict__ outp)
{
  constexpr int BM = 128, BN = 64, BK = 32, LDA = 40, LDB = 40;
  __shared__ __align__(16) _Float16 sA[BM * LDA];
  __shared__ __align__(16) _Float16 sB[BN * LDB];

  const int m0 = blockIdx.y * BM;
  const int n0 = blockIdx.x * BN;
  const int tid  = threadIdx.x;
  const int wave = tid >> 5;
  const int wm = (wave >> 1) * 32, wn = (wave & 1) * 32;

  v4f ra[4], rb[2];
  auto load_tiles = [&](int k0) {
#pragma unroll
    for (int i = 0; i < 4; ++i) {
      int ch = tid + i * 256;
      int r = ch >> 3, c = (ch & 7) * 4;
      ra[i] = *(const v4f*)(X + (size_t)(m0 + r) * DMODEL + k0 + c);
    }
#pragma unroll
    for (int i = 0; i < 2; ++i) {
      int ch = tid + i * 256;
      int r = ch >> 3, c = (ch & 7) * 4;
      rb[i] = *(const v4f*)(W + (size_t)(n0 + r) * DMODEL + k0 + c);
    }
  };
  auto store_tiles = [&]() {
#pragma unroll
    for (int i = 0; i < 4; ++i) {
      int ch = tid + i * 256;
      int r = ch >> 3, c = (ch & 7) * 4;
      *(v4h*)(sA + r * LDA + c) = __builtin_convertvector(ra[i], v4h);
    }
#pragma unroll
    for (int i = 0; i < 2; ++i) {
      int ch = tid + i * 256;
      int r = ch >> 3, c = (ch & 7) * 4;
      *(v4h*)(sB + r * LDB + c) = __builtin_convertvector(rb[i], v4h);
    }
  };

  v8f acc[2][2] = {};
  load_tiles(0);
  for (int k0 = 0; k0 < DMODEL; k0 += BK) {
    store_tiles();
    __syncthreads();
    if (k0 + BK < DMODEL) load_tiles(k0 + BK);
    v16h a0 = frag_a(sA, LDA, wm, 0);
    v16h a1 = frag_a(sA, LDA, wm + 16, 0);
    v16h b0 = frag_b(sB, LDB, wn, 0);
    v16h b1 = frag_b(sB, LDB, wn + 16, 0);
    acc[0][0] = wmma_f16(a0, b0, acc[0][0]);
    acc[0][1] = wmma_f16(a0, b1, acc[0][1]);
    acc[1][0] = wmma_f16(a1, b0, acc[1][0]);
    acc[1][1] = wmma_f16(a1, b1, acc[1][1]);
    __syncthreads();
  }

  const int lane = tid & 31;
#pragma unroll
  for (int im = 0; im < 2; ++im)
#pragma unroll
    for (int in = 0; in < 2; ++in) {
      int n  = n0 + wn + in * 16 + (lane & 15);
      int mb = m0 + wm + im * 16 + ((lane >> 4) << 3);
      float bb = bias[n];
#pragma unroll
      for (int r = 0; r < 8; ++r) {
        size_t off = (size_t)(mb + r) * DMODEL + n;
        outp[off] = acc[im][in][r] + bb + residual[off];
      }
    }
}

// ---------------------------------------------------------------------------
// Kernel 6: LayerNorm per row of 1024. grid MROWS, block 256
// ---------------------------------------------------------------------------
__global__ __launch_bounds__(256) void ln_kernel(
    const float* __restrict__ x, const float* __restrict__ g,
    const float* __restrict__ bta, float* __restrict__ outp)
{
  const size_t row = blockIdx.x;
  const float* p = x + row * DMODEL;
  const int tid = threadIdx.x;

  float vals[4];
  float s = 0.f;
#pragma unroll
  for (int i = 0; i < 4; ++i) { vals[i] = p[tid + i * 256]; s += vals[i]; }
#pragma unroll
  for (int off = 16; off > 0; off >>= 1) s += __shfl_xor(s, off, 32);

  __shared__ float red1[8];
  __shared__ float red2[8];
  if ((tid & 31) == 0) red1[tid >> 5] = s;
  __syncthreads();
  float mu = ((red1[0] + red1[1]) + (red1[2] + red1[3]) +
              (red1[4] + red1[5]) + (red1[6] + red1[7])) * (1.0f / DMODEL);

  float vs = 0.f;
#pragma unroll
  for (int i = 0; i < 4; ++i) { float d = vals[i] - mu; vs += d * d; }
#pragma unroll
  for (int off = 16; off > 0; off >>= 1) vs += __shfl_xor(vs, off, 32);
  if ((tid & 31) == 0) red2[tid >> 5] = vs;
  __syncthreads();
  float var = ((red2[0] + red2[1]) + (red2[2] + red2[3]) +
               (red2[4] + red2[5]) + (red2[6] + red2[7])) * (1.0f / DMODEL);
  float inv = rsqrtf(var + 1e-5f);

#pragma unroll
  for (int i = 0; i < 4; ++i) {
    int n = tid + i * 256;
    outp[row * DMODEL + n] = (vals[i] - mu) * inv * g[n] + bta[n];
  }
}

// ---------------------------------------------------------------------------
extern "C" void kernel_launch(void* const* d_in, const int* in_sizes, int n_in,
                              void* d_out, int out_size, void* d_ws, size_t ws_size,
                              hipStream_t stream) {
  const float* q    = (const float*)d_in[0];
  const float* k    = (const float*)d_in[1];
  const float* v    = (const float*)d_in[2];
  const unsigned char* mask = (const unsigned char*)d_in[3];
  const float* w_q  = (const float*)d_in[4];
  const float* b_q  = (const float*)d_in[5];
  const float* w_k  = (const float*)d_in[6];
  const float* b_k  = (const float*)d_in[7];
  const float* w_v  = (const float*)d_in[8];
  const float* b_v  = (const float*)d_in[9];
  const float* w_fc = (const float*)d_in[10];
  const float* b_fc = (const float*)d_in[11];
  const float* ln_g = (const float*)d_in[12];
  const float* ln_b = (const float*)d_in[13];

  float* out_x    = (float*)d_out;                            // [8,1024,1024]
  float* out_attn = out_x + (size_t)MROWS * DMODEL;           // [8,16,1024,1024]

  // ws: qh f16 [0,16M) | kh f16 [16M,32M) | vt f16 [32M,48M)
  //     o f32 [48M,80M) | x f32 reuses [0,32M) (qh/kh dead by then)
  char* ws = (char*)d_ws;
  _Float16* qh = (_Float16*)(ws);
  _Float16* kh = (_Float16*)(ws + (size_t)(16 << 20));
  _Float16* vt = (_Float16*)(ws + (size_t)(32 << 20));
  float*    o  = (float*)   (ws + (size_t)(48 << 20));
  float*    xb = (float*)   (ws);

  proj_qkv_kernel<<<dim3(DMODEL / 64, MROWS / 128, 3), 256, 0, stream>>>(
      q, k, v, w_q, w_k, w_v, b_q, b_k, b_v, qh, kh, vt);
  scores_kernel<<<dim3(SEQ / 64, SEQ / 128, BATCH * NHEAD), 256, 0, stream>>>(
      qh, kh, mask, out_attn);
  softmax_kernel<<<dim3(BATCH * NHEAD * SEQ), 256, 0, stream>>>(out_attn);
  attnv_kernel<<<dim3(1, SEQ / 128, BATCH * NHEAD), 256, 0, stream>>>(
      out_attn, vt, o);
  fc_kernel<<<dim3(DMODEL / 64, MROWS / 128), 256, 0, stream>>>(
      o, w_fc, b_fc, q, xb);
  ln_kernel<<<dim3(MROWS), 256, 0, stream>>>(xb, ln_g, ln_b, out_x);
}
